// TopoCycleResidualNodeAlpha_49641232007243
// MI455X (gfx1250) — compile-verified, matrix operand order checked by reading, not run on hardware
//
#include <hip/hip_runtime.h>
#include <hip/hip_bf16.h>

// Problem constants (match reference)
#define BB   8
#define NN   2048
#define EE   8192
#define TT   4096
#define HID  128

#define LDSK 40          // LDS row stride in bf16 elems (80B, keeps b128 frags 16B-aligned)

typedef unsigned short ushort_t;
typedef __attribute__((ext_vector_type(16))) __bf16 v16bf;
typedef __attribute__((ext_vector_type(8)))  float  v8f;

__device__ __forceinline__ ushort_t f2bf(float f) {
    union { float f; unsigned u; } v; v.f = f;
    unsigned r = v.u + 0x7FFFu + ((v.u >> 16) & 1u);   // round-to-nearest-even
    return (ushort_t)(r >> 16);
}

__device__ __forceinline__ v8f zero8() {
    v8f z;
#pragma unroll
    for (int i = 0; i < 8; ++i) z[i] = 0.0f;
    return z;
}

// Async DMA: 16B global -> LDS, tracked by ASYNCcnt (no VGPR data path).
// LDS operand is the low 32 bits of the generic pointer (ISA 10.2 aperture rule).
__device__ __forceinline__ void async_b128(const ushort_t* g, ushort_t* l) {
    unsigned lds = (unsigned)(unsigned long long)l;
    asm volatile("global_load_async_to_lds_b128 %0, %1, off"
                 :: "v"(lds), "v"(g) : "memory");
}
__device__ __forceinline__ void wait_async() {
#if __has_builtin(__builtin_amdgcn_s_wait_asynccnt)
    __builtin_amdgcn_s_wait_asynccnt(0);
#else
    asm volatile("s_wait_asynccnt 0" ::: "memory");
#endif
}

// Per-lane fragment slice from LDS tile (row-major, LDSK stride).
// Lanes 0-15: K=0..7 & 16..23 of row m; lanes 16-31: K=8..15 & 24..31 (ISA 7.12.2).
__device__ __forceinline__ v16bf load_frag(const ushort_t* rowbase, int khalf) {
    union { v16bf v; uint4 q[2]; } u;
    u.q[0] = *(const uint4*)(rowbase + khalf * 8);
    u.q[1] = *(const uint4*)(rowbase + 16 + khalf * 8);
    return u.v;
}

// ================= pre-passes =================

__global__ __launch_bounds__(256) void k_diag_b1(const float* __restrict__ B1g,
                                                 float* __restrict__ L1) {
    int e = blockIdx.x * 256 + threadIdx.x;
    float acc = 0.f;
    for (int n = 0; n < NN; ++n) { float v = B1g[(size_t)n * EE + e]; acc += v * v; }
    L1[e] = acc;
}

__global__ __launch_bounds__(256) void k_diag_b2(const float* __restrict__ B2g,
                                                 float* __restrict__ L1) {
    __shared__ float red[256];
    int e = blockIdx.x, tid = threadIdx.x;
    float acc = 0.f;
    for (int t = tid; t < TT; t += 256) { float v = B2g[(size_t)e * TT + t]; acc += v * v; }
    red[tid] = acc; __syncthreads();
    for (int s = 128; s > 0; s >>= 1) { if (tid < s) red[tid] += red[tid + s]; __syncthreads(); }
    if (tid == 0) L1[e] += red[0];
}

// B1 fp32 -> bf16 row-major (b1b [N,E]) + transposed (b1t [E,N])
__global__ __launch_bounds__(256) void k_cvt_b1(const float* __restrict__ B1g,
                                                ushort_t* __restrict__ b1b,
                                                ushort_t* __restrict__ b1t) {
    __shared__ ushort_t tile[32][33];
    int ex = blockIdx.x * 32, ny = blockIdx.y * 32;
    int tx = threadIdx.x & 31, ty = threadIdx.x >> 5;
#pragma unroll
    for (int i = 0; i < 4; ++i) {
        int n = ny + ty + i * 8;
        ushort_t v = f2bf(B1g[(size_t)n * EE + ex + tx]);
        b1b[(size_t)n * EE + ex + tx] = v;
        tile[ty + i * 8][tx] = v;
    }
    __syncthreads();
#pragma unroll
    for (int i = 0; i < 4; ++i) {
        int e = ex + ty + i * 8;
        b1t[(size_t)e * NN + ny + tx] = tile[tx][ty + i * 8];
    }
}

// H fp32 -> bf16 transpose (hbT [B,HID,N])
__global__ __launch_bounds__(256) void k_cvt_h(const float* __restrict__ Hg,
                                               ushort_t* __restrict__ hbT) {
    __shared__ ushort_t tile[32][33];
    int hx = blockIdx.x * 32, ny = blockIdx.y * 32, b = blockIdx.z;
    int tx = threadIdx.x & 31, ty = threadIdx.x >> 5;
#pragma unroll
    for (int i = 0; i < 4; ++i) {
        int n = ny + ty + i * 8;
        tile[ty + i * 8][tx] = f2bf(Hg[((size_t)b * NN + n) * HID + hx + tx]);
    }
    __syncthreads();
#pragma unroll
    for (int i = 0; i < 4; ++i) {
        int h = hx + ty + i * 8;
        hbT[((size_t)b * HID + h) * NN + ny + tx] = tile[tx][ty + i * 8];
    }
}

__global__ __launch_bounds__(256) void k_cvt_w(const float* __restrict__ Wg,
                                               ushort_t* __restrict__ wb) {
    int i = blockIdx.x * 256 + threadIdx.x;
    wb[i] = f2bf(Wg[i]);
}

// ======== GEMM1: Etopo[b,e,h] = L1[e] * sum_n b1t[e,n] * hbT[b,h,n] ========
// Block tile 256(M) x 128(N); each wave owns 32x128 (2 A-frags, 16 WMMA / K-step).
__global__ __launch_bounds__(256) void k_gemm1(const ushort_t* __restrict__ b1t,
                                               const ushort_t* __restrict__ hbT,
                                               const float* __restrict__ L1,
                                               ushort_t* __restrict__ Etopo) {
    __shared__ ushort_t At[2][256 * LDSK];
    __shared__ ushort_t Bt[2][128 * LDSK];
    const int b = blockIdx.y, eBase = blockIdx.x * 256;
    const int tid = threadIdx.x, wave = tid >> 5, lane = tid & 31;
    const int m16 = lane & 15, khalf = lane >> 4, mrow = khalf * 8, col = m16;

    v8f acc0[8], acc1[8];
#pragma unroll
    for (int j = 0; j < 8; ++j) { acc0[j] = zero8(); acc1[j] = zero8(); }

    auto stage = [&](int k0, int p) {
#pragma unroll
        for (int i = 0; i < 4; ++i) {                     // A: 256x32 = 1024 uint4
            int idx = i * 256 + tid;
            int r = idx >> 2, q = idx & 3;
            async_b128(b1t + (size_t)(eBase + r) * NN + k0 + q * 8,
                       &At[p][r * LDSK + q * 8]);
        }
#pragma unroll
        for (int i = 0; i < 2; ++i) {                     // B: 128x32 = 512 uint4
            int idx = i * 256 + tid;
            int r = idx >> 2, q = idx & 3;
            async_b128(hbT + ((size_t)b * HID + r) * NN + k0 + q * 8,
                       &Bt[p][r * LDSK + q * 8]);
        }
    };

    stage(0, 0);
    wait_async();
    __syncthreads();
    int p = 0;
    for (int k0 = 32; k0 <= NN; k0 += 32) {
        if (k0 < NN) stage(k0, 1 - p);                    // DMA next tile during compute
        v16bf af0 = load_frag(&At[p][(wave * 32 + m16) * LDSK], khalf);
        v16bf af1 = load_frag(&At[p][(wave * 32 + 16 + m16) * LDSK], khalf);
#pragma unroll
        for (int j = 0; j < 8; ++j) {
            v16bf bf = load_frag(&Bt[p][(j * 16 + m16) * LDSK], khalf);
            acc0[j] = __builtin_amdgcn_wmma_f32_16x16x32_bf16(
                false, af0, false, bf, (short)0, acc0[j], false, false);
            acc1[j] = __builtin_amdgcn_wmma_f32_16x16x32_bf16(
                false, af1, false, bf, (short)0, acc1[j], false, false);
        }
        if (k0 < NN) wait_async();
        __syncthreads();
        p ^= 1;
    }

#pragma unroll
    for (int s = 0; s < 2; ++s) {
        v8f* acc = s ? acc1 : acc0;
        float scale[8];
#pragma unroll
        for (int r = 0; r < 8; ++r) scale[r] = L1[eBase + wave * 32 + s * 16 + mrow + r];
#pragma unroll
        for (int j = 0; j < 8; ++j) {
            int h = j * 16 + col;
#pragma unroll
            for (int r = 0; r < 8; ++r) {
                int e = eBase + wave * 32 + s * 16 + mrow + r;
                Etopo[((size_t)b * EE + e) * HID + h] = f2bf(acc[j][r] * scale[r]);
            }
        }
    }
}

// ======== GEMM2: dhb[b,n,h] = bf16( sum_e b1b[n,e] * Etopo[b,e,h] ) ========
// A tile (256x32): async DMA. B tile: transposed bf16 staging through VGPRs.
__global__ __launch_bounds__(256) void k_gemm2(const ushort_t* __restrict__ b1b,
                                               const ushort_t* __restrict__ Etopo,
                                               ushort_t* __restrict__ dhb) {
    __shared__ ushort_t At[2][256 * LDSK];
    __shared__ ushort_t Bt[2][128 * LDSK];
    const int b = blockIdx.y, nBase = blockIdx.x * 256;
    const int tid = threadIdx.x, wave = tid >> 5, lane = tid & 31;
    const int m16 = lane & 15, khalf = lane >> 4, mrow = khalf * 8, col = m16;

    v8f acc0[8], acc1[8];
#pragma unroll
    for (int j = 0; j < 8; ++j) { acc0[j] = zero8(); acc1[j] = zero8(); }

    auto stageA = [&](int k0, int p) {
#pragma unroll
        for (int i = 0; i < 4; ++i) {
            int idx = i * 256 + tid;
            int r = idx >> 2, q = idx & 3;
            async_b128(b1b + (size_t)(nBase + r) * EE + k0 + q * 8,
                       &At[p][r * LDSK + q * 8]);
        }
    };
    ushort_t sB[16];
    auto loadB = [&](int k0) {
#pragma unroll
        for (int i = 0; i < 16; ++i) {
            int idx = i * 256 + tid;
            int h = idx & 127, ek = idx >> 7;
            sB[i] = Etopo[((size_t)b * EE + k0 + ek) * HID + h];   // coalesced over h
        }
    };
    auto storeB = [&](int p) {
#pragma unroll
        for (int i = 0; i < 16; ++i) {
            int idx = i * 256 + tid;
            int h = idx & 127, ek = idx >> 7;
            Bt[p][h * LDSK + ek] = sB[i];
        }
    };

    stageA(0, 0); loadB(0); storeB(0);
    wait_async();
    __syncthreads();
    int p = 0;
    for (int k0 = 32; k0 <= EE; k0 += 32) {
        if (k0 < EE) { stageA(k0, 1 - p); loadB(k0); }
        v16bf af0 = load_frag(&At[p][(wave * 32 + m16) * LDSK], khalf);
        v16bf af1 = load_frag(&At[p][(wave * 32 + 16 + m16) * LDSK], khalf);
#pragma unroll
        for (int j = 0; j < 8; ++j) {
            v16bf bf = load_frag(&Bt[p][(j * 16 + m16) * LDSK], khalf);
            acc0[j] = __builtin_amdgcn_wmma_f32_16x16x32_bf16(
                false, af0, false, bf, (short)0, acc0[j], false, false);
            acc1[j] = __builtin_amdgcn_wmma_f32_16x16x32_bf16(
                false, af1, false, bf, (short)0, acc1[j], false, false);
        }
        if (k0 < EE) { storeB(1 - p); wait_async(); }
        __syncthreads();
        p ^= 1;
    }

#pragma unroll
    for (int s = 0; s < 2; ++s) {
        v8f* acc = s ? acc1 : acc0;
#pragma unroll
        for (int j = 0; j < 8; ++j) {
            int h = j * 16 + col;
#pragma unroll
            for (int r = 0; r < 8; ++r) {
                int n = nBase + wave * 32 + s * 16 + mrow + r;
                dhb[((size_t)b * NN + n) * HID + h] = f2bf(acc[j][r]);
            }
        }
    }
}

// ======== GEMM3: out = H + alpha * relu(dhb @ W^T) ========
__global__ __launch_bounds__(256) void k_gemm3(const ushort_t* __restrict__ dhb,
                                               const ushort_t* __restrict__ wb,
                                               const float* __restrict__ Hg,
                                               const float* __restrict__ alpha_raw,
                                               float* __restrict__ out) {
    __shared__ ushort_t At[128 * LDSK];
    __shared__ ushort_t Wt[128 * LDSK];
    const int b = blockIdx.y, nBase = blockIdx.x * 128;
    const int tid = threadIdx.x, wave = tid >> 5, lane = tid & 31;
    const int m16 = lane & 15, khalf = lane >> 4, mrow = khalf * 8, col = m16;
    const int r0 = (tid * 2) >> 2,     q0 = (tid * 2) & 3;
    const int r1 = (tid * 2 + 1) >> 2, q1 = (tid * 2 + 1) & 3;

    v8f acc[8];
#pragma unroll
    for (int j = 0; j < 8; ++j) acc[j] = zero8();

    for (int k0 = 0; k0 < HID; k0 += 32) {
        async_b128(dhb + ((size_t)b * NN + nBase + r0) * HID + k0 + q0 * 8, &At[r0 * LDSK + q0 * 8]);
        async_b128(dhb + ((size_t)b * NN + nBase + r1) * HID + k0 + q1 * 8, &At[r1 * LDSK + q1 * 8]);
        async_b128(wb + (size_t)r0 * HID + k0 + q0 * 8, &Wt[r0 * LDSK + q0 * 8]);
        async_b128(wb + (size_t)r1 * HID + k0 + q1 * 8, &Wt[r1 * LDSK + q1 * 8]);
        wait_async();
        __syncthreads();
        v16bf af = load_frag(&At[(wave * 16 + m16) * LDSK], khalf);
#pragma unroll
        for (int j = 0; j < 8; ++j) {
            v16bf bf = load_frag(&Wt[(j * 16 + m16) * LDSK], khalf);
            acc[j] = __builtin_amdgcn_wmma_f32_16x16x32_bf16(
                false, af, false, bf, (short)0, acc[j], false, false);
        }
        __syncthreads();
    }

    float alpha = 0.05f / (1.0f + __expf(-alpha_raw[0]));
#pragma unroll
    for (int j = 0; j < 8; ++j) {
        int o = j * 16 + col;
#pragma unroll
        for (int r = 0; r < 8; ++r) {
            int n = nBase + wave * 16 + mrow + r;
            size_t off = ((size_t)b * NN + n) * HID + o;
            float v = acc[j][r];
            v = v > 0.0f ? v : 0.0f;
            out[off] = Hg[off] + alpha * v;
        }
    }
}

extern "C" void kernel_launch(void* const* d_in, const int* in_sizes, int n_in,
                              void* d_out, int out_size, void* d_ws, size_t ws_size,
                              hipStream_t stream) {
    const float* H   = (const float*)d_in[0];
    const float* B1  = (const float*)d_in[1];
    const float* B2  = (const float*)d_in[2];
    const float* W   = (const float*)d_in[3];
    const float* a_r = (const float*)d_in[4];
    float* out = (float*)d_out;
    (void)in_sizes; (void)n_in; (void)out_size; (void)ws_size;

    char* ws = (char*)d_ws;
    size_t off = 0;
    auto take = [&](size_t bytes) { void* p = ws + off; off = (off + bytes + 255) & ~(size_t)255; return p; };
    float*    L1    = (float*)   take((size_t)EE * 4);
    ushort_t* b1b   = (ushort_t*)take((size_t)NN * EE * 2);
    ushort_t* b1t   = (ushort_t*)take((size_t)EE * NN * 2);
    ushort_t* hbT   = (ushort_t*)take((size_t)BB * HID * NN * 2);
    ushort_t* wb    = (ushort_t*)take((size_t)HID * HID * 2);
    ushort_t* Etopo = (ushort_t*)take((size_t)BB * EE * HID * 2);
    ushort_t* dhb   = (ushort_t*)take((size_t)BB * NN * HID * 2);

    k_diag_b1<<<EE / 256, 256, 0, stream>>>(B1, L1);
    k_diag_b2<<<EE, 256, 0, stream>>>(B2, L1);

    dim3 gb1(EE / 32, NN / 32);
    k_cvt_b1<<<gb1, 256, 0, stream>>>(B1, b1b, b1t);
    dim3 gh(HID / 32, NN / 32, BB);
    k_cvt_h<<<gh, 256, 0, stream>>>(H, hbT);
    k_cvt_w<<<(HID * HID) / 256, 256, 0, stream>>>(W, wb);

    dim3 g1(EE / 256, BB);
    k_gemm1<<<g1, 256, 0, stream>>>(b1t, hbT, L1, Etopo);
    dim3 g2(NN / 256, BB);
    k_gemm2<<<g2, 256, 0, stream>>>(b1b, Etopo, dhb);
    dim3 g3(NN / 128, BB);
    k_gemm3<<<g3, 256, 0, stream>>>(dhb, wb, H, a_r, out);
}
